// TICALPlugin_58025008169588
// MI455X (gfx1250) — compile-verified, matrix operand order checked by reading, not run on hardware
//
#include <hip/hip_runtime.h>
#include <hip/hip_bf16.h>
#include <math.h>

typedef __attribute__((ext_vector_type(16))) _Float16 v16h;
typedef __attribute__((ext_vector_type(8)))  float    v8f;

namespace cfg {
constexpr int B = 128, L = 512, H = 336, D = 256, TXT = 768;
constexpr int K = 16, S = 4, EMB = 128;
constexpr int RAWC = K * (3 + S);      // 112
constexpr int GIN  = 2 * D + TXT;      // 1280
constexpr int G1   = 512;
constexpr int FK   = 3 * D;            // 768
constexpr int SHIFT = H / 6;           // 56
constexpr float H_SCALE = 0.15f, EPS = 0.05f;
constexpr int   ITERS = 30;
constexpr float ALPHA = 1.0f, BETA = 0.5f, GW = 0.5f, TAU = 0.07f;
constexpr float LMB_MSE = 1.0f, LMB_COT = 0.1f, LMB_DELTA = 0.1f,
                LMB_ENT = 0.01f, LMB_TV = 0.01f;
constexpr float PI_F = 3.14159265358979323846f;
}

__device__ __forceinline__ float softplusf(float x) {
  return fmaxf(x, 0.f) + log1pf(__expf(-fabsf(x)));
}
__device__ __forceinline__ float sigmoidf(float x) {
  return 1.f / (1.f + __expf(-x));
}

// ---- WMMA operand helpers -------------------------------------------------
__device__ __forceinline__ float4 ld4(const float* __restrict__ p) {
  return *reinterpret_cast<const float4*>(p);
}
__device__ __forceinline__ float4 sub4(float4 a, float4 b) {
  return make_float4(a.x - b.x, a.y - b.y, a.z - b.z, a.w - b.w);
}
__device__ __forceinline__ float4 d2c(float4 a, float4 b, float4 c) {
  return make_float4(a.x - 2.f * b.x + c.x, a.y - 2.f * b.y + c.y,
                     a.z - 2.f * b.z + c.z, a.w - 2.f * b.w + c.w);
}
__device__ __forceinline__ void pk8(v16h& v, int base, float4 a, float4 b) {
  v[base + 0] = (_Float16)a.x; v[base + 1] = (_Float16)a.y;
  v[base + 2] = (_Float16)a.z; v[base + 3] = (_Float16)a.w;
  v[base + 4] = (_Float16)b.x; v[base + 5] = (_Float16)b.y;
  v[base + 6] = (_Float16)b.z; v[base + 7] = (_Float16)b.w;
}
// B operand: column n fixed per lane, 8 K-values at k0.. and k0+16..
template<int N>
__device__ __forceinline__ void fillB(v16h& bv, const float* __restrict__ Bp, int k0) {
#pragma unroll
  for (int t = 0; t < 8; ++t) {
    bv[t]     = (_Float16)Bp[(size_t)(k0 + t) * N];
    bv[8 + t] = (_Float16)Bp[(size_t)(k0 + 16 + t) * N];
  }
}

// ---------------------------------------------------------------------------
// WMMA GEMM: C[M,N] = act(A[M,Kd] @ Bm[Kd,N] + bias[N]).  One wave per 16x16
// tile, v_wmma_f32_16x16x32_f16.  Lane layout per ISA 7.12.2: lanes 0-15 hold
// K {0..7,16..23}, lanes 16-31 hold K {8..15,24..31} of each 32-wide slab.
// Compile-time N/Kd -> immediate-offset addressing, no bounds checks in the
// hot loop (Kd%32==16 tail = low-half load + zero high half, once).
// ACT: 0=none, 1=relu, 2=sigmoid.
// ---------------------------------------------------------------------------
template<int N, int Kd, int ACT>
__global__ void wmma_gemm_t(const float* __restrict__ A, const float* __restrict__ Bm,
                            const float* __restrict__ bias, float* __restrict__ Cout,
                            int M)
{
  static_assert((Kd % 32 == 0) || (Kd % 32 == 16), "Kd tail must be 0 or 16");
  int wave = blockIdx.x * (blockDim.x >> 5) + (threadIdx.x >> 5);
  constexpr int ntx = N >> 4;
  int total = (M >> 4) * ntx;
  if (wave >= total) return;                 // wave-uniform guard
  int tm = wave / ntx, tn = wave % ntx;
  int m0 = tm << 4, n0 = tn << 4;
  int lane = threadIdx.x & 31;
  int hiG = lane >> 4, idx = lane & 15;
  const float* Arow = A + (size_t)(m0 + idx) * Kd + hiG * 8;
  const float* Bp = Bm + n0 + idx + (size_t)hiG * 8 * N;
  v8f acc = {};
  constexpr int KMAIN = Kd & ~31;
  for (int k0 = 0; k0 < KMAIN; k0 += 32) {
    v16h av, bv;
    pk8(av, 0, ld4(Arow + k0), ld4(Arow + k0 + 4));
    pk8(av, 8, ld4(Arow + k0 + 16), ld4(Arow + k0 + 20));
    fillB<N>(bv, Bp, k0);
    acc = __builtin_amdgcn_wmma_f32_16x16x32_f16(false, av, false, bv,
                                                 (short)0, acc, false, false);
  }
  if constexpr ((Kd & 31) == 16) {           // 16-wide tail: high K-half is OOB
    v16h av, bv;
    pk8(av, 0, ld4(Arow + KMAIN), ld4(Arow + KMAIN + 4));
#pragma unroll
    for (int t = 0; t < 8; ++t) {
      av[8 + t] = (_Float16)0.f;
      bv[t] = (_Float16)Bp[(size_t)(KMAIN + t) * N];
      bv[8 + t] = (_Float16)0.f;
    }
    acc = __builtin_amdgcn_wmma_f32_16x16x32_f16(false, av, false, bv,
                                                 (short)0, acc, false, false);
  }
  float bb = bias ? bias[n0 + idx] : 0.f;
#pragma unroll
  for (int j = 0; j < 8; ++j) {              // C: VGPR j -> rows j / j+8
    int row = m0 + j + hiG * 8;
    float v = acc[j] + bb;
    if constexpr (ACT == 1) v = fmaxf(v, 0.f);
    if constexpr (ACT == 2) v = sigmoidf(v);
    Cout[(size_t)row * N + (n0 + idx)] = v;
  }
}

// ---------------------------------------------------------------------------
// f_pos = [y, diff(y), diff2(y)] @ W_fp + b_fp via WMMA.  The 3*D concat is
// generated in-register: sections are 256-wide and slab-aligned, so each
// K-slab's section is loop-uniform; h-boundary handling is branch-free via
// clamped rows (d1 = y[h]-y[hm1]; d2 = y[h]-2y[hm1]+y[hm2] reproduces the
// reference's prepend semantics for h=0,1 exactly).
// ---------------------------------------------------------------------------
__global__ void wmma_fproj(const float* __restrict__ Y, const float* __restrict__ Wfp,
                           const float* __restrict__ bfp, float* __restrict__ Fout)
{
  using namespace cfg;
  constexpr int N = EMB;                     // 128
  constexpr int ntx = N / 16;
  constexpr int total = (B * H / 16) * ntx;
  int wave = blockIdx.x * (blockDim.x >> 5) + (threadIdx.x >> 5);
  if (wave >= total) return;
  int tm = wave / ntx, tn = wave % ntx;
  int m0 = tm << 4, n0 = tn << 4;
  int lane = threadIdx.x & 31;
  int hiG = lane >> 4, idx = lane & 15;
  int row = m0 + idx;
  int b = row / H, h = row % H;              // H%16==0: tile stays in one b
  int hm1 = (h >= 1) ? h - 1 : 0;
  int hm2 = (h >= 2) ? h - 2 : 0;
  const float* p0 = Y + ((size_t)b * H + h) * D + hiG * 8;
  const float* p1 = Y + ((size_t)b * H + hm1) * D + hiG * 8;
  const float* p2 = Y + ((size_t)b * H + hm2) * D + hiG * 8;
  const float* Bp = Wfp + n0 + idx + (size_t)hiG * 8 * N;
  v8f acc = {};
  // section 0: y
  for (int k0 = 0; k0 < D; k0 += 32) {
    v16h av, bv;
    pk8(av, 0, ld4(p0 + k0), ld4(p0 + k0 + 4));
    pk8(av, 8, ld4(p0 + k0 + 16), ld4(p0 + k0 + 20));
    fillB<N>(bv, Bp, k0);
    acc = __builtin_amdgcn_wmma_f32_16x16x32_f16(false, av, false, bv,
                                                 (short)0, acc, false, false);
  }
  // section 1: d1 = y[h] - y[hm1]
  for (int k0 = 0; k0 < D; k0 += 32) {
    v16h av, bv;
    pk8(av, 0, sub4(ld4(p0 + k0), ld4(p1 + k0)),
               sub4(ld4(p0 + k0 + 4), ld4(p1 + k0 + 4)));
    pk8(av, 8, sub4(ld4(p0 + k0 + 16), ld4(p1 + k0 + 16)),
               sub4(ld4(p0 + k0 + 20), ld4(p1 + k0 + 20)));
    fillB<N>(bv, Bp, D + k0);
    acc = __builtin_amdgcn_wmma_f32_16x16x32_f16(false, av, false, bv,
                                                 (short)0, acc, false, false);
  }
  // section 2: d2 = y[h] - 2*y[hm1] + y[hm2]
  for (int k0 = 0; k0 < D; k0 += 32) {
    v16h av, bv;
    pk8(av, 0, d2c(ld4(p0 + k0), ld4(p1 + k0), ld4(p2 + k0)),
               d2c(ld4(p0 + k0 + 4), ld4(p1 + k0 + 4), ld4(p2 + k0 + 4)));
    pk8(av, 8, d2c(ld4(p0 + k0 + 16), ld4(p1 + k0 + 16), ld4(p2 + k0 + 16)),
               d2c(ld4(p0 + k0 + 20), ld4(p1 + k0 + 20), ld4(p2 + k0 + 20)));
    fillB<N>(bv, Bp, 2 * D + k0);
    acc = __builtin_amdgcn_wmma_f32_16x16x32_f16(false, av, false, bv,
                                                 (short)0, acc, false, false);
  }
  float bb = bfp[n0 + idx];
#pragma unroll
  for (int j = 0; j < 8; ++j) {
    int r2 = m0 + j + hiG * 8;
    Fout[(size_t)r2 * N + (n0 + idx)] = acc[j] + bb;
  }
}

// ---------------------------------------------------------------------------
// Small elementwise / reduction kernels
// ---------------------------------------------------------------------------
__global__ void zero_f32(float* p, int n)
{
  int i = blockIdx.x * blockDim.x + threadIdx.x;
  if (i < n) p[i] = 0.f;
}

__global__ void params_kernel(const float* __restrict__ raw, float* __restrict__ mu,
                              float* __restrict__ sig, float* __restrict__ amp,
                              float* __restrict__ sw, float* __restrict__ loss)
{
  using namespace cfg;
  int i = blockIdx.x * blockDim.x + threadIdx.x;   // b*K + k
  if (i >= B * K) return;
  const float* r = raw + (size_t)i * (3 + S);
  mu[i]  = sigmoidf(r[0]);
  sig[i] = softplusf(r[1]) * H_SCALE + 1e-3f;
  amp[i] = softplusf(r[2]);
  float mx = r[3];
  for (int s = 1; s < S; ++s) mx = fmaxf(mx, r[3 + s]);
  float e[S], tot = 0.f;
  for (int s = 0; s < S; ++s) { e[s] = __expf(r[3 + s] - mx); tot += e[s]; }
  float ent = 0.f;
  for (int s = 0; s < S; ++s) {
    float p = e[s] / tot;
    sw[(size_t)i * S + s] = p;
    float pc = fmaxf(p, 1e-8f);
    ent += pc * __logf(pc);
  }
  atomicAdd(loss, LMB_ENT * (-ent) / (float)(B * K));
}

__global__ void tv_kernel(const float* __restrict__ mu, float* __restrict__ loss)
{
  using namespace cfg;
  int i = blockIdx.x * blockDim.x + threadIdx.x;   // b*(K-1) + j
  if (i >= B * (K - 1)) return;
  int b = i / (K - 1), j = i % (K - 1);
  float d = fabsf(mu[b * K + j + 1] - mu[b * K + j]);
  atomicAdd(loss, LMB_TV * d / (float)(B * (K - 1)));
}

__global__ void kappa_kernel(const float* __restrict__ mu, const float* __restrict__ sig,
                             const float* __restrict__ amp, const float* __restrict__ sw,
                             float* __restrict__ kappa)
{
  using namespace cfg;
  int i = blockIdx.x * blockDim.x + threadIdx.x;   // (b*K+k)*H + h
  if (i >= B * K * H) return;
  int h = i % H;
  int bk = i / H;
  float t = (h + 0.5f) / (float)H;
  float z = (t - mu[bk]) / sig[bk];
  float az = fabsf(z);
  float in_supp = (az <= 1.f) ? 1.f : 0.f;
  float b0 = __expf(-0.5f * z * z);
  float b1 = __expf(-az);
  float b2 = fmaxf(1.f - az, 0.f);
  float zc = fminf(fmaxf(z, -1.f), 1.f);
  float b3 = 0.5f * (1.f + __cosf(PI_F * zc)) * in_supp;
  const float* w = sw + (size_t)bk * S;
  kappa[i] = amp[bk] * (w[0] * b0 + w[1] * b1 + w[2] * b2 + w[3] * b3);
}

__global__ void rsum_kernel(const float* __restrict__ kappa, float* __restrict__ rvec)
{
  using namespace cfg;
  int i = blockIdx.x * blockDim.x + threadIdx.x;   // b*K + k
  if (i >= B * K) return;
  const float* row = kappa + (size_t)i * H;
  float s = 0.f;
  for (int h = 0; h < H; ++h) s += row[h];
  rvec[i] = s + 1e-6f;
}

__global__ void rnorm_kernel(float* __restrict__ rvec)
{
  using namespace cfg;
  int b = blockIdx.x * blockDim.x + threadIdx.x;
  if (b >= B) return;
  float s = 0.f;
  for (int k = 0; k < K; ++k) s += rvec[b * K + k];
  for (int k = 0; k < K; ++k) rvec[b * K + k] /= s;
}

__global__ void ytsc_kernel(const float* __restrict__ kappa, float* __restrict__ ytsc)
{
  using namespace cfg;
  int i = blockIdx.x * blockDim.x + threadIdx.x;   // b*H + h
  if (i >= B * H) return;
  int b = i / H, h = i % H;
  float s = 0.f;
  for (int k = 0; k < K; ++k) s += kappa[((size_t)b * K + k) * H + h];
  ytsc[i] = s;
}

__global__ void tsg_kernel(const float* __restrict__ enc, float* __restrict__ tsg)
{
  using namespace cfg;
  int i = blockIdx.x * blockDim.x + threadIdx.x;   // b*D + d
  if (i >= B * D) return;
  int b = i / D, d = i % D;
  const float* p = enc + (size_t)b * L * D + d;
  float s = 0.f;
  for (int l = 0; l < L; ++l) s += p[(size_t)l * D];
  tsg[i] = s / (float)L;
}

__global__ void gin_kernel(const float* __restrict__ tsg, const float* __restrict__ text,
                           const float* __restrict__ var_proj, float* __restrict__ gin)
{
  using namespace cfg;
  int i = blockIdx.x * blockDim.x + threadIdx.x;   // b*GIN + j
  if (i >= B * GIN) return;
  int b = i / GIN, j = i % GIN;
  float v;
  if (j < D)            v = tsg[b * D + j];
  else if (j < D + TXT) v = text[(size_t)b * TXT + (j - D)];
  else                  v = var_proj[j - D - TXT];
  gin[i] = v;
}

__global__ void mse_kernel(const float* __restrict__ y_res, const float* __restrict__ y_fut,
                           const float* __restrict__ ytsc, const float* __restrict__ gate,
                           const float* __restrict__ W_tir, const float* __restrict__ b_tir,
                           const float* __restrict__ var_proj, float* __restrict__ loss)
{
  using namespace cfg;
  __shared__ float red[256];
  size_t total = (size_t)B * H * D;
  size_t i = (size_t)blockIdx.x * blockDim.x + threadIdx.x;
  float acc = 0.f;
  for (; i < total; i += (size_t)gridDim.x * blockDim.x) {
    int d = (int)(i % D);
    size_t bh = i / D;
    int h = (int)(bh % H);
    int b = (int)(bh / H);
    float yr = y_res[i];
    float ytir = (ytsc[b * H + h] * W_tir[d] + b_tir[d]) * var_proj[d];
    float yh = yr + GW * gate[b * D + d] * (ytir - yr);
    float df = yh - y_fut[i];
    acc += df * df;
  }
  red[threadIdx.x] = acc;
  __syncthreads();
  for (int s = blockDim.x / 2; s > 0; s >>= 1) {
    if ((int)threadIdx.x < s) red[threadIdx.x] += red[threadIdx.x + s];
    __syncthreads();
  }
  if (threadIdx.x == 0) atomicAdd(loss, LMB_MSE * red[0] / (float)total);
}

__global__ void knorm_kernel(const float* __restrict__ kemb, float* __restrict__ kn)
{
  using namespace cfg;
  int i = blockIdx.x * blockDim.x + threadIdx.x;   // b*K + k
  if (i >= B * K) return;
  const float* src = kemb + (size_t)i * EMB;
  float s = 0.f;
  for (int d = 0; d < EMB; ++d) s += src[d] * src[d];
  float inv = 1.f / (sqrtf(s) + 1e-8f);
  float* dst = kn + (size_t)i * EMB;
  for (int d = 0; d < EMB; ++d) dst[d] = src[d] * inv;
}

__global__ void fnorm_kernel(const float* __restrict__ fpos, float* __restrict__ fnorm)
{
  using namespace cfg;
  int i = blockIdx.x * blockDim.x + threadIdx.x;   // b*H + h
  if (i >= B * H) return;
  const float* src = fpos + (size_t)i * EMB;
  float s = 0.f;
  for (int d = 0; d < EMB; ++d) s += src[d] * src[d];
  fnorm[i] = sqrtf(s) + 1e-8f;
}

__global__ void cost_kernel(const float* __restrict__ kn, const float* __restrict__ fpos,
                            const float* __restrict__ fnorm, const float* __restrict__ mu,
                            float* __restrict__ Cmat, float* __restrict__ Kmat)
{
  using namespace cfg;
  int i = blockIdx.x * blockDim.x + threadIdx.x;   // (b*K+k)*H + h
  if (i >= B * K * H) return;
  int h = i % H;
  int bk = i / H;
  int b = bk / K;
  const float* kr = kn + (size_t)bk * EMB;
  const float* fr = fpos + ((size_t)b * H + h) * EMB;
  float dot = 0.f;
#pragma unroll 8
  for (int d = 0; d < EMB; ++d) dot += kr[d] * fr[d];
  float cs = dot / fnorm[b * H + h];
  float t = (h + 0.5f) / (float)H;
  float cv = ALPHA * (1.f - cs) + BETA * fmaxf(mu[bk] - t, 0.f);
  Cmat[i] = cv;
  Kmat[i] = __expf(-cv / EPS);
}

// One workgroup (512 thr = 16 waves) per batch element; K-matrix in LDS.
__global__ void __launch_bounds__(512)
sinkhorn_kernel(const float* __restrict__ Cg, const float* __restrict__ Kg,
                const float* __restrict__ rvec, float* __restrict__ Pi,
                float* __restrict__ wsum, float* __restrict__ loss)
{
  using namespace cfg;
  __shared__ float Ks[K * H];     // 21.5 KB
  __shared__ float us[K];
  __shared__ float vs[H];
  __shared__ float cotred[K];
  int b = blockIdx.x;
  const float* Kb = Kg + (size_t)b * K * H;
  for (int i = threadIdx.x; i < K * H; i += blockDim.x) Ks[i] = Kb[i];
  if ((int)threadIdx.x < K) us[threadIdx.x] = 1.f;
  for (int i = threadIdx.x; i < H; i += blockDim.x) vs[i] = 1.f;
  __syncthreads();
  int wave = threadIdx.x >> 5;     // == k (16 waves)
  int lane = threadIdx.x & 31;
  for (int it = 0; it < ITERS; ++it) {
    float p = 0.f;
    for (int h = lane; h < H; h += 32) p += Ks[wave * H + h] * vs[h];
    for (int off = 16; off > 0; off >>= 1) p += __shfl_xor(p, off);
    if (lane == 0) us[wave] = rvec[b * K + wave] / (p + 1e-9f);
    __syncthreads();
    for (int h = threadIdx.x; h < H; h += blockDim.x) {
      float q = 0.f;
#pragma unroll
      for (int k = 0; k < K; ++k) q += Ks[k * H + h] * us[k];
      vs[h] = (1.f / (float)H) / (q + 1e-9f);
    }
    __syncthreads();
  }
  const float* Cb = Cg + (size_t)b * K * H;
  float wacc = 0.f, cot = 0.f;
  for (int h = lane; h < H; h += 32) {
    float pi = us[wave] * Ks[wave * H + h] * vs[h];
    Pi[(size_t)b * K * H + wave * H + h] = pi;
    wacc += pi;
    cot += Cb[wave * H + h] * pi;
  }
  for (int off = 16; off > 0; off >>= 1) {
    wacc += __shfl_xor(wacc, off);
    cot  += __shfl_xor(cot, off);
  }
  if (lane == 0) { wsum[b * K + wave] = wacc; cotred[wave] = cot; }
  __syncthreads();
  if (threadIdx.x == 0) {
    float s = 0.f;
    for (int k = 0; k < K; ++k) s += cotred[k];
    atomicAdd(loss, LMB_COT * s / (float)B);
  }
}

// One workgroup (128 thr) per (b,k): transport-weighted aggregation + 3-way
// contrastive log-softmax.
__global__ void __launch_bounds__(128)
delta_kernel(const float* __restrict__ kn, const float* __restrict__ fpos,
             const float* __restrict__ Pi, const float* __restrict__ wsum,
             const int* __restrict__ perm, float* __restrict__ loss)
{
  using namespace cfg;
  __shared__ float red[128];
  __shared__ float res[6];
  int bk = blockIdx.x;
  int b = bk / K;
  int d = threadIdx.x;            // 0..127 (EMB)
  const float* Pr = Pi + (size_t)bk * H;
  float wden = wsum[bk] + 1e-9f;
  const float* F = fpos + (size_t)b * H * EMB;
  int pd = perm[d];
  float a0 = 0.f, a1 = 0.f, a2 = 0.f;
  for (int h = 0; h < H; ++h) {
    float w = Pr[h] / wden;
    int hs = h - SHIFT; if (hs < 0) hs += H;   // roll(f_pos, +SHIFT)
    a0 += w * F[(size_t)h * EMB + d];
    a1 += w * F[(size_t)hs * EMB + d];
    a2 += w * F[(size_t)h * EMB + pd];
  }
  float kd = kn[(size_t)bk * EMB + d];
  float vals[6] = { a0 * a0, a1 * a1, a2 * a2, kd * a0, kd * a1, kd * a2 };
  for (int q = 0; q < 6; ++q) {
    red[d] = vals[q];
    __syncthreads();
    for (int s = 64; s > 0; s >>= 1) {
      if (d < s) red[d] += red[d + s];
      __syncthreads();
    }
    if (d == 0) res[q] = red[0];
    __syncthreads();
  }
  if (d == 0) {
    float l0 = (res[3] / (sqrtf(res[0]) + 1e-8f)) / TAU;
    float l1 = (res[4] / (sqrtf(res[1]) + 1e-8f)) / TAU;
    float l2 = (res[5] / (sqrtf(res[2]) + 1e-8f)) / TAU;
    float m = fmaxf(l0, fmaxf(l1, l2));
    float lse = m + __logf(__expf(l0 - m) + __expf(l1 - m) + __expf(l2 - m));
    atomicAdd(loss, -(l0 - lse) * LMB_DELTA / (float)(B * K));
  }
}

// ---------------------------------------------------------------------------
extern "C" void kernel_launch(void* const* d_in, const int* in_sizes, int n_in,
                              void* d_out, int out_size, void* d_ws, size_t ws_size,
                              hipStream_t stream)
{
  using namespace cfg;
  (void)in_sizes; (void)n_in; (void)ws_size;
  const float* y_res    = (const float*)d_in[0];
  const float* text_emb = (const float*)d_in[1];
  const float* enc_out  = (const float*)d_in[2];
  const float* y_future = (const float*)d_in[3];
  const int*   perm     = (const int*)d_in[4];
  const float* W_t2k    = (const float*)d_in[5];
  const float* b_t2k    = (const float*)d_in[6];
  const float* W_kemb   = (const float*)d_in[7];
  const float* b_kemb   = (const float*)d_in[8];
  const float* W_tir    = (const float*)d_in[9];
  const float* b_tir    = (const float*)d_in[10];
  const float* var_proj = (const float*)d_in[11];
  const float* W_g1     = (const float*)d_in[12];
  const float* b_g1     = (const float*)d_in[13];
  const float* W_g2     = (const float*)d_in[14];
  const float* b_g2     = (const float*)d_in[15];
  const float* W_fp     = (const float*)d_in[16];
  const float* b_fp     = (const float*)d_in[17];

  float* ws = (float*)d_ws;
  float* raw  = ws; ws += (size_t)B * RAWC;
  float* mu   = ws; ws += B * K;
  float* sig  = ws; ws += B * K;
  float* amp  = ws; ws += B * K;
  float* sw   = ws; ws += (size_t)B * K * S;
  float* kap  = ws; ws += (size_t)B * K * H;
  float* rvec = ws; ws += B * K;
  float* ytsc = ws; ws += (size_t)B * H;
  float* tsg  = ws; ws += (size_t)B * D;
  float* gin  = ws; ws += (size_t)B * GIN;
  float* hid  = ws; ws += (size_t)B * G1;
  float* gate = ws; ws += (size_t)B * D;
  float* kemb = ws; ws += (size_t)B * K * EMB;
  float* kn   = ws; ws += (size_t)B * K * EMB;
  float* fpos = ws; ws += (size_t)B * H * EMB;
  float* fnrm = ws; ws += (size_t)B * H;
  float* Cmat = ws; ws += (size_t)B * K * H;
  float* Kmat = ws; ws += (size_t)B * K * H;
  float* Pim  = ws; ws += (size_t)B * K * H;
  float* wsum = ws; ws += B * K;
  float* loss = (float*)d_out;

  auto blocks_for = [](int M, int N) { return ((M / 16) * (N / 16) + 3) / 4; };

  zero_f32<<<1, 64, 0, stream>>>(loss, out_size);

  // text -> kernel params (WMMA GEMM)
  wmma_gemm_t<RAWC, TXT, 0><<<blocks_for(B, RAWC), 128, 0, stream>>>(
      text_emb, W_t2k, b_t2k, raw, B);
  params_kernel<<<(B * K + 127) / 128, 128, 0, stream>>>(raw, mu, sig, amp, sw, loss);
  tv_kernel<<<(B * (K - 1) + 127) / 128, 128, 0, stream>>>(mu, loss);
  kappa_kernel<<<(B * K * H + 255) / 256, 256, 0, stream>>>(mu, sig, amp, sw, kap);
  rsum_kernel<<<(B * K + 127) / 128, 128, 0, stream>>>(kap, rvec);
  rnorm_kernel<<<1, B, 0, stream>>>(rvec);
  ytsc_kernel<<<(B * H + 255) / 256, 256, 0, stream>>>(kap, ytsc);

  // gated fusion + MSE
  tsg_kernel<<<(B * D + 255) / 256, 256, 0, stream>>>(enc_out, tsg);
  gin_kernel<<<(B * GIN + 255) / 256, 256, 0, stream>>>(tsg, text_emb, var_proj, gin);
  wmma_gemm_t<G1, GIN, 1><<<blocks_for(B, G1), 128, 0, stream>>>(
      gin, W_g1, b_g1, hid, B);                             // relu
  wmma_gemm_t<D, G1, 2><<<blocks_for(B, D), 128, 0, stream>>>(
      hid, W_g2, b_g2, gate, B);                            // sigmoid
  mse_kernel<<<2048, 256, 0, stream>>>(y_res, y_future, ytsc, gate,
                                       W_tir, b_tir, var_proj, loss);

  // embeddings + future projection (big WMMA GEMM)
  wmma_gemm_t<EMB, H, 0><<<blocks_for(B * K, EMB), 128, 0, stream>>>(
      kap, W_kemb, b_kemb, kemb, B * K);                    // Kd=336 -> 16-tail
  wmma_fproj<<<blocks_for(B * H, EMB), 128, 0, stream>>>(y_future, W_fp, b_fp, fpos);
  knorm_kernel<<<(B * K + 127) / 128, 128, 0, stream>>>(kemb, kn);
  fnorm_kernel<<<(B * H + 255) / 256, 256, 0, stream>>>(fpos, fnrm);

  // cost, Sinkhorn OT, contrastive
  cost_kernel<<<(B * K * H + 255) / 256, 256, 0, stream>>>(kn, fpos, fnrm, mu, Cmat, Kmat);
  sinkhorn_kernel<<<B, 512, 0, stream>>>(Cmat, Kmat, rvec, Pim, wsum, loss);
  delta_kernel<<<B * K, 128, 0, stream>>>(kn, fpos, Pim, wsum, perm, loss);
}
